// SlotAttention_73538430042529
// MI455X (gfx1250) — matrix operand order repaired and checked by reading, not compiled, not run on hardware
//
#include <hip/hip_runtime.h>
#include <hip/hip_bf16.h>
#include <cmath>

// ---------------- problem constants ----------------
constexpr int B_  = 16;
constexpr int N_  = 4096;
constexpr int D_  = 512;
constexpr int NS_ = 16;
constexpr int H_  = 512;      // hidden dim of MLP
constexpr int G3_ = 3 * D_;   // 1536 gate width

typedef _Float16 half_t;
typedef __attribute__((ext_vector_type(16))) _Float16 v16h;
typedef __attribute__((ext_vector_type(8)))  _Float16 v8h;
typedef __attribute__((ext_vector_type(8)))  float    v8f;

// ---------------- WMMA fragment helpers (wave32, 16x16x32 f16) ----------------
// A-matrix 16x32 (MxK) f16 layout: lanes 0-15 -> M=lane, halves 0..7 = K 0..7,
// halves 8..15 = K 16..23 ; lanes 16-31 -> M=lane-16, K 8..15 and 24..31.
__device__ __forceinline__ v16h load_a_frag(const half_t* __restrict__ A, int lda,
                                            int m0, int k0) {
  int lane = threadIdx.x & 31;
  const half_t* p = A + (size_t)(m0 + (lane & 15)) * lda + k0 + ((lane >> 4) << 3);
  v8h lo = *(const v8h*)(p);
  v8h hi = *(const v8h*)(p + 16);
  v16h r;
#pragma unroll
  for (int i = 0; i < 8; ++i) { r[i] = lo[i]; r[i + 8] = hi[i]; }
  return r;
}

// B-matrix 32x16 (KxN) f16 layout, sourced from a row-major (N x K) matrix Bt
// (i.e. we compute A * Bt^T): lanes 0-15 -> N=lane, K 0..7 and 16..23;
// lanes 16-31 -> N=lane-16, K 8..15 and 24..31.
__device__ __forceinline__ v16h load_bt_frag(const half_t* __restrict__ Bt, int ldb,
                                             int n0, int k0) {
  int lane = threadIdx.x & 31;
  const half_t* p = Bt + (size_t)(n0 + (lane & 15)) * ldb + k0 + ((lane >> 4) << 3);
  v8h lo = *(const v8h*)(p);
  v8h hi = *(const v8h*)(p + 16);
  v16h r;
#pragma unroll
  for (int i = 0; i < 8; ++i) { r[i] = lo[i]; r[i + 8] = hi[i]; }
  return r;
}

// C/D 16x16 f32: lanes 0-15 -> N=lane, VGPR i -> M=i ; lanes 16-31 -> M=8+i.
__device__ __forceinline__ void store_c_frag_nt(float* __restrict__ C, int ldc,
                                                int m0, int n0, v8f c) {
  int lane = threadIdx.x & 31;
  int n  = n0 + (lane & 15);
  int mb = m0 + ((lane >> 4) << 3);
#pragma unroll
  for (int i = 0; i < 8; ++i)
    __builtin_nontemporal_store(c[i], &C[(size_t)(mb + i) * ldc + n]);
}

// ---------------- generic batched WMMA GEMM:  C = epi(alpha * A @ Bt^T) ----------------
// flags: bit0 = relu
// NOTE: no explicit prefetch in the K loop — round-2 feedback showed the guarded
// prefetch broke the compiler's clause pipelining (s_wait_loadcnt 0x0 before each
// WMMA instead of 0x3). The plain loop gets 4 loads in flight past each WMMA.
__global__ __launch_bounds__(256) void wmma_gemm_bt(
    const half_t* __restrict__ A, long long sA, int lda,
    const half_t* __restrict__ Bt, long long sB, int ldb,
    int tilesM, int tilesN, int K,
    const float* __restrict__ bias, float alpha,
    const float* __restrict__ rowscale, long long sRS,
    float* Cf, long long sCf,
    half_t* Ch, long long sCh, int ldc,
    const float* __restrict__ resid, long long sR,
    int flags) {
  int wave = threadIdx.x >> 5;
  int lane = threadIdx.x & 31;
  int t = blockIdx.x * 8 + wave;
  if (t >= tilesM * tilesN) return;
  int tm = t / tilesN, tn = t % tilesN;
  int bb = blockIdx.y;
  A  += (size_t)bb * sA;
  Bt += (size_t)bb * sB;
  if (Cf)       Cf       += (size_t)bb * sCf;
  if (Ch)       Ch       += (size_t)bb * sCh;
  if (rowscale) rowscale += (size_t)bb * sRS;
  if (resid)    resid    += (size_t)bb * sR;

  int m0 = tm * 16, n0 = tn * 16;
  v8f acc = {};
  for (int k0 = 0; k0 < K; k0 += 32) {
    v16h a = load_a_frag(A, lda, m0, k0);
    v16h b = load_bt_frag(Bt, ldb, n0, k0);
    acc = __builtin_amdgcn_wmma_f32_16x16x32_f16(false, a, false, b,
                                                 (short)0, acc, false, false);
  }
  int n  = n0 + (lane & 15);
  int mb = m0 + ((lane >> 4) << 3);
  float bn = bias ? bias[n] : 0.0f;
#pragma unroll
  for (int i = 0; i < 8; ++i) {
    int m = mb + i;
    float v = acc[i] * alpha;
    if (rowscale) v *= rowscale[m];
    v += bn;
    if (flags & 1) v = v > 0.0f ? v : 0.0f;
    size_t idx = (size_t)m * ldc + n;
    if (resid) v += resid[idx];
    if (Cf) Cf[idx] = v;
    if (Ch) Ch[idx] = (half_t)v;
  }
}

// ---------------- LayerNorm over D=512, 256 threads/row ----------------
// mode 0: out[row*D + d] (row-major f16). mode 1: out[(b*D + d)*N + j] (transposed per batch).
// Input rows are read exactly once (k_inp / v_inp / slots) -> non-temporal loads.
__global__ __launch_bounds__(256) void ln_rows_kernel(
    const float* __restrict__ in, const float* __restrict__ g,
    const float* __restrict__ be, half_t* __restrict__ out, int mode, int Ntok) {
  int row = blockIdx.x, tid = threadIdx.x;
  const float* x = in + (size_t)row * D_;
  float v0 = __builtin_nontemporal_load(&x[tid]);
  float v1 = __builtin_nontemporal_load(&x[tid + 256]);
  __shared__ float red[256];
  red[tid] = v0 + v1;
  __syncthreads();
  for (int s = 128; s > 0; s >>= 1) { if (tid < s) red[tid] += red[tid + s]; __syncthreads(); }
  float mu = red[0] * (1.0f / D_);
  __syncthreads();
  float d0 = v0 - mu, d1 = v1 - mu;
  red[tid] = d0 * d0 + d1 * d1;
  __syncthreads();
  for (int s = 128; s > 0; s >>= 1) { if (tid < s) red[tid] += red[tid + s]; __syncthreads(); }
  float rstd = rsqrtf(red[0] * (1.0f / D_) + 1e-5f);
  float y0 = d0 * rstd * g[tid]       + be[tid];
  float y1 = d1 * rstd * g[tid + 256] + be[tid + 256];
  if (mode == 0) {
    out[(size_t)row * D_ + tid]       = (half_t)y0;
    out[(size_t)row * D_ + tid + 256] = (half_t)y1;
  } else {
    int b = row / Ntok, j = row % Ntok;
    half_t* o = out + (size_t)b * D_ * Ntok + j;
    o[(size_t)tid * Ntok]         = (half_t)y0;
    o[(size_t)(tid + 256) * Ntok] = (half_t)y1;
  }
}

// ---------------- softmax over the NS=16 slots for each (b, j) ----------------
__global__ void softmax_slots_kernel(const float* __restrict__ dots,
                                     float* __restrict__ attn,
                                     half_t* __restrict__ attnh) {
  int idx = blockIdx.x * blockDim.x + threadIdx.x;
  if (idx >= B_ * N_) return;
  int b = idx / N_, j = idx % N_;
  const float* dp = dots + (size_t)b * NS_ * N_ + j;
  float x[NS_], m = -3.4e38f;
#pragma unroll
  for (int i = 0; i < NS_; ++i) { x[i] = dp[(size_t)i * N_]; m = fmaxf(m, x[i]); }
  float s = 0.0f;
#pragma unroll
  for (int i = 0; i < NS_; ++i) { x[i] = expf(x[i] - m); s += x[i]; }
  float inv = 1.0f / s;
  float*  ap = attn  + (size_t)b * NS_ * N_ + j;
  half_t* ah = attnh + (size_t)b * NS_ * N_ + j;
#pragma unroll
  for (int i = 0; i < NS_; ++i) {
    float a = x[i] * inv + 1e-8f;
    ap[(size_t)i * N_] = a;          // re-read by rowsum/feature -> keep cached
    ah[(size_t)i * N_] = (half_t)a;  // re-read by updates GEMM   -> keep cached
  }
}

// ---------------- per (b, slot) row sum over j, store reciprocal ----------------
__global__ __launch_bounds__(256) void rowsum_inv_kernel(const float* __restrict__ attn,
                                                         float* __restrict__ inv_sum) {
  int row = blockIdx.x;
  const float* p = attn + (size_t)row * N_;
  float s = 0.0f;
  for (int j = threadIdx.x; j < N_; j += 256) s += p[j];
  __shared__ float red[256];
  red[threadIdx.x] = s;
  __syncthreads();
  for (int k = 128; k > 0; k >>= 1) { if (threadIdx.x < k) red[threadIdx.x] += red[threadIdx.x + k]; __syncthreads(); }
  if (threadIdx.x == 0) inv_sum[row] = 1.0f / red[0];
}

// ---------------- GRU pointwise update (torch gate order r,z,n) ----------------
__global__ void gru_update_kernel(const float* __restrict__ gx, const float* __restrict__ gh,
                                  float* __restrict__ s32, half_t* __restrict__ s16) {
  int idx = blockIdx.x * blockDim.x + threadIdx.x;
  if (idx >= B_ * NS_ * D_) return;
  int r = idx / D_, d = idx % D_;
  const float* gxr = gx + (size_t)r * G3_;
  const float* ghr = gh + (size_t)r * G3_;
  float rg = 1.0f / (1.0f + expf(-(gxr[d]          + ghr[d])));
  float zg = 1.0f / (1.0f + expf(-(gxr[D_ + d]     + ghr[D_ + d])));
  float ng = tanhf(gxr[2 * D_ + d] + rg * ghr[2 * D_ + d]);
  float h  = s32[idx];
  float o  = (1.0f - zg) * ng + zg * h;
  s32[idx] = o;
  s16[idx] = (half_t)o;
}

// ---------------- misc elementwise kernels ----------------
__global__ void f32_to_f16_kernel(const float* __restrict__ in, half_t* __restrict__ out, int n) {
  int i = blockIdx.x * blockDim.x + threadIdx.x;
  if (i < n) out[i] = (half_t)in[i];
}
__global__ void bcast_slots_kernel(const float* __restrict__ mu,
                                   float* __restrict__ s32, half_t* __restrict__ s16) {
  int i = blockIdx.x * blockDim.x + threadIdx.x;
  if (i < B_ * NS_ * D_) { float v = mu[i % (NS_ * D_)]; s32[i] = v; s16[i] = (half_t)v; }
}
__global__ void prm_pad_kernel(const float* __restrict__ prompts, half_t* __restrict__ prmT) {
  int i = blockIdx.x * blockDim.x + threadIdx.x;
  if (i >= D_ * 32) return;
  int d = i / 32, n = i % 32;
  prmT[i] = (half_t)(n < NS_ ? prompts[n * D_ + d] : 0.0f);
}
__global__ void slotsT_pad_kernel(const float* __restrict__ s32, half_t* __restrict__ sT) {
  int i = blockIdx.x * blockDim.x + threadIdx.x;
  if (i >= B_ * D_ * 32) return;
  int b = i / (D_ * 32), r = i % (D_ * 32), d = r / 32, n = r % 32;
  sT[i] = (half_t)(n < NS_ ? s32[((size_t)b * NS_ + n) * D_ + d] : 0.0f);
}
__global__ void copy_f32_nt_kernel(const float* __restrict__ in, float* __restrict__ out, int n) {
  int i = blockIdx.x * blockDim.x + threadIdx.x;
  if (i < n) __builtin_nontemporal_store(in[i], &out[i]);
}

// ---------------- final feature einsum: out[b,p,d] = sum_n M[b,n,d]*attn[b,n,p] ----------------
// K=16 zero-padded to 32; one 16(p) x 16(d) tile per wave, two WMMAs (prompt & slot).
// 403 MB of write-once output -> non-temporal stores so the L2 keeps kh/vT hot.
__global__ __launch_bounds__(256) void feature_kernel(
    const float* __restrict__ attn, const half_t* __restrict__ prmT,
    const half_t* __restrict__ slotsT,
    float* __restrict__ outP, float* __restrict__ outS1, float* __restrict__ outS2) {
  int wave = threadIdx.x >> 5, lane = threadIdx.x & 31;
  int t = blockIdx.x * 8 + wave;
  const int PT = N_ / 16, DT = D_ / 16;             // 256, 32
  int b = t / (PT * DT), rem = t % (PT * DT);
  int p0 = (rem / DT) * 16, d0 = (rem % DT) * 16;

  v16h af;
  int m  = p0 + (lane & 15);
  int kb = (lane >> 4) << 3;                        // slot index base 0 or 8
  const float* ap = attn + (size_t)b * NS_ * N_ + m;
#pragma unroll
  for (int h = 0; h < 8; ++h) af[h] = (half_t)ap[(size_t)(kb + h) * N_];
#pragma unroll
  for (int h = 8; h < 16; ++h) af[h] = (half_t)0.0f; // K=16..31 zero pad

  v16h bp = load_bt_frag(prmT, 32, d0, 0);
  v16h bs = load_bt_frag(slotsT + (size_t)b * D_ * 32, 32, d0, 0);
  v8f z = {};
  v8f cp = __builtin_amdgcn_wmma_f32_16x16x32_f16(false, af, false, bp, (short)0, z, false, false);
  v8f cs = __builtin_amdgcn_wmma_f32_16x16x32_f16(false, af, false, bs, (short)0, z, false, false);

  store_c_frag_nt(outP  + (size_t)b * N_ * D_, D_, p0, d0, cp);
  store_c_frag_nt(outS1 + (size_t)b * N_ * D_, D_, p0, d0, cs);
  store_c_frag_nt(outS2 + (size_t)b * N_ * D_, D_, p0, d0, cs);
}

// ---------------- host side ----------------
extern "C" void kernel_launch(void* const* d_in, const int* in_sizes, int n_in,
                              void* d_out, int out_size, void* d_ws, size_t ws_size,
                              hipStream_t stream) {
  const float* k_inp    = (const float*)d_in[2];
  const float* v_inp    = (const float*)d_in[3];
  const float* slots_mu = (const float*)d_in[4];
  const float* prompts  = (const float*)d_in[5];
  const float* Wq   = (const float*)d_in[6];
  const float* bq   = (const float*)d_in[7];
  const float* W_ih = (const float*)d_in[8];
  const float* b_ih = (const float*)d_in[9];
  const float* W_hh = (const float*)d_in[10];
  const float* b_hh = (const float*)d_in[11];
  const float* W1   = (const float*)d_in[12];
  const float* b1   = (const float*)d_in[13];
  const float* W2   = (const float*)d_in[14];
  const float* b2   = (const float*)d_in[15];
  const float* gk = (const float*)d_in[16]; const float* bk = (const float*)d_in[17];
  const float* gv = (const float*)d_in[18]; const float* bv = (const float*)d_in[19];
  const float* gs = (const float*)d_in[20]; const float* bs = (const float*)d_in[21];
  const float* gff = (const float*)d_in[22]; const float* bff = (const float*)d_in[23];

  float* out = (float*)d_out;
  const size_t OFF_SLOTS = 0;
  const size_t OFF_ATTN  = (size_t)B_ * NS_ * D_;                 // 131072
  const size_t OFF_PF    = OFF_ATTN + (size_t)B_ * NS_ * N_;      // 1179648
  const size_t OFF_SF1   = OFF_PF  + (size_t)B_ * N_ * D_;
  const size_t OFF_SF2   = OFF_SF1 + (size_t)B_ * N_ * D_;
  float* attn_f32 = out + OFF_ATTN;

  // ---- workspace bump allocator (total ~150 MB) ----
  char* w = (char*)d_ws;
  auto alloc = [&](size_t bytes) -> char* {
    char* p = w; w += (bytes + 255) & ~(size_t)255; return p;
  };
  half_t* kh      = (half_t*)alloc((size_t)B_ * N_ * D_ * 2);   // LN(k), row-major
  half_t* vT      = (half_t*)alloc((size_t)B_ * D_ * N_ * 2);   // LN(v), transposed per batch
  half_t* Wq_h    = (half_t*)alloc((size_t)D_ * D_ * 2);
  half_t* Wih_h   = (half_t*)alloc((size_t)G3_ * D_ * 2);
  half_t* Whh_h   = (half_t*)alloc((size_t)G3_ * D_ * 2);
  half_t* W1_h    = (half_t*)alloc((size_t)H_ * D_ * 2);
  half_t* W2_h    = (half_t*)alloc((size_t)D_ * H_ * 2);
  half_t* prmT    = (half_t*)alloc((size_t)D_ * 32 * 2);
  half_t* slotsT  = (half_t*)alloc((size_t)B_ * D_ * 32 * 2);
  float*  s32     = (float*) alloc((size_t)B_ * NS_ * D_ * 4);
  half_t* s16     = (half_t*)alloc((size_t)B_ * NS_ * D_ * 2);
  half_t* sh      = (half_t*)alloc((size_t)B_ * NS_ * D_ * 2);  // LN(slots)
  half_t* qh      = (half_t*)alloc((size_t)B_ * NS_ * D_ * 2);
  float*  dots    = (float*) alloc((size_t)B_ * NS_ * N_ * 4);
  half_t* attnh   = (half_t*)alloc((size_t)B_ * NS_ * N_ * 2);
  float*  inv_sum = (float*) alloc((size_t)B_ * NS_ * 4);
  half_t* upd_h   = (half_t*)alloc((size_t)B_ * NS_ * D_ * 2);
  float*  gxb     = (float*) alloc((size_t)B_ * NS_ * G3_ * 4);
  float*  ghb     = (float*) alloc((size_t)B_ * NS_ * G3_ * 4);
  half_t* ffh     = (half_t*)alloc((size_t)B_ * NS_ * D_ * 2);
  half_t* hid_h   = (half_t*)alloc((size_t)B_ * NS_ * H_ * 2);

  const float scale = 1.0f / sqrtf((float)D_);

  // ---- one-time prep ----
  f32_to_f16_kernel<<<(D_*D_ + 255)/256, 256, 0, stream>>>(Wq, Wq_h, D_*D_);
  f32_to_f16_kernel<<<(G3_*D_ + 255)/256, 256, 0, stream>>>(W_ih, Wih_h, G3_*D_);
  f32_to_f16_kernel<<<(G3_*D_ + 255)/256, 256, 0, stream>>>(W_hh, Whh_h, G3_*D_);
  f32_to_f16_kernel<<<(H_*D_ + 255)/256, 256, 0, stream>>>(W1, W1_h, H_*D_);
  f32_to_f16_kernel<<<(D_*H_ + 255)/256, 256, 0, stream>>>(W2, W2_h, D_*H_);
  prm_pad_kernel<<<(D_*32 + 255)/256, 256, 0, stream>>>(prompts, prmT);

  ln_rows_kernel<<<B_*N_, 256, 0, stream>>>(k_inp, gk, bk, kh, 0, N_);  // k f16 row-major
  ln_rows_kernel<<<B_*N_, 256, 0, stream>>>(v_inp, gv, bv, vT, 1, N_);  // v f16 transposed
  bcast_slots_kernel<<<(B_*NS_*D_ + 255)/256, 256, 0, stream>>>(slots_mu, s32, s16);

  // ---- 3 slot-attention iterations ----
  for (int it = 0; it < 3; ++it) {
    // s = LN(slots)
    ln_rows_kernel<<<B_*NS_, 256, 0, stream>>>(s32, gs, bs, sh, 0, NS_);
    // q = s @ Wq^T + bq   (M=256, N=512, K=512)
    wmma_gemm_bt<<<dim3(64, 1), 256, 0, stream>>>(
        sh, 0, D_, Wq_h, 0, D_, 16, 32, D_, bq, 1.0f,
        nullptr, 0, nullptr, 0, qh, 0, D_, nullptr, 0, 0);
    // dots[b] = q[b] @ k[b]^T * scale   (batched: M=16, N=4096, K=512)
    wmma_gemm_bt<<<dim3(32, B_), 256, 0, stream>>>(
        qh, (long long)NS_*D_, D_, kh, (long long)N_*D_, D_, 1, N_/16, D_,
        nullptr, scale, nullptr, 0, dots, (long long)NS_*N_, nullptr, 0, N_,
        nullptr, 0, 0);
    // softmax over slots (axis=1) + EPS -> attn (f32 into d_out slice) + f16 copy
    softmax_slots_kernel<<<(B_*N_ + 255)/256, 256, 0, stream>>>(dots, attn_f32, attnh);
    // 1 / sum_j attn
    rowsum_inv_kernel<<<B_*NS_, 256, 0, stream>>>(attn_f32, inv_sum);
    // updates[b] = diag(inv_sum) @ attn[b] @ v[b]   (M=16, N=512, K=4096)
    wmma_gemm_bt<<<dim3(4, B_), 256, 0, stream>>>(
        attnh, (long long)NS_*N_, N_, vT, (long long)D_*N_, N_, 1, D_/16, N_,
        nullptr, 1.0f, inv_sum, NS_, nullptr, 0, upd_h, (long long)NS_*D_, D_,
        nullptr, 0, 0);
    // GRU gates: gx = x @ W_ih^T + b_ih ; gh = h @ W_hh^T + b_hh  (M=256, N=1536, K=512)
    wmma_gemm_bt<<<dim3(192, 1), 256, 0, stream>>>(
        upd_h, 0, D_, Wih_h, 0, D_, 16, G3_/16, D_, b_ih, 1.0f,
        nullptr, 0, gxb, 0, nullptr, 0, G3_, nullptr, 0, 0);
    wmma_gemm_bt<<<dim3(192, 1), 256, 0, stream>>>(
        s16, 0, D_, Whh_h, 0, D_, 16, G3_/16, D_, b_hh, 1.0f,
        nullptr, 0, ghb, 0, nullptr, 0, G3_, nullptr, 0, 0);
    gru_update_kernel<<<(B_*NS_*D_ + 255)/256, 256, 0, stream>>>(gxb, ghb, s32, s16);
    // ff = LN(slots) ; slots += relu(ff @ W1^T + b1) @ W2^T + b2
    ln_rows_kernel<<<B_*NS_, 256, 0, stream>>>(s32, gff, bff, ffh, 0, NS_);
    wmma_gemm_bt<<<dim3(64, 1), 256, 0, stream>>>(
        ffh, 0, D_, W1_h, 0, D_, 16, H_/16, D_, b1, 1.0f,
        nullptr, 0, nullptr, 0, hid_h, 0, H_, nullptr, 0, /*relu*/1);
    wmma_gemm_bt<<<dim3(64, 1), 256, 0, stream>>>(
        hid_h, 0, H_, W2_h, 0, H_, 16, D_/16, H_, b2, 1.0f,
        nullptr, 0, s32, 0, s16, 0, D_, /*resid*/s32, 0, 0);
  }

  // ---- outputs ----
  copy_f32_nt_kernel<<<(B_*NS_*D_ + 255)/256, 256, 0, stream>>>(s32, out + OFF_SLOTS, B_*NS_*D_);
  slotsT_pad_kernel<<<(B_*D_*32 + 255)/256, 256, 0, stream>>>(s32, slotsT);
  feature_kernel<<<(B_*(N_/16)*(D_/16))/8, 256, 0, stream>>>(
      attn_f32, prmT, slotsT,
      out + OFF_PF, out + OFF_SF1, out + OFF_SF2);
}
